// TransformerEncoderCustom_56547539419803
// MI455X (gfx1250) — matrix operand order repaired and checked by reading, not compiled, not run on hardware
//
#include <hip/hip_runtime.h>
#include <hip/hip_bf16.h>

// ---------------------------------------------------------------------------
// Problem constants (from the reference)
// ---------------------------------------------------------------------------
#define BB   16
#define LL   512
#define DD   512
#define HH   8
#define DHH  64
#define DFFN 2048
#define NLAY 4
#define MM   (BB * LL)         // 8192 rows of activations
#define WINDOW 32
#define NEGB (-1.0e9f)

// LDS row stride in uints: 20 -> 80B rows (16B aligned for async B128 writes)
// and conflict-free across 16 rows (gcd(20,64)=4, cycle length 16).
#define LDS_STRIDE 20

typedef __attribute__((ext_vector_type(16))) __bf16 v16bf;
typedef __attribute__((ext_vector_type(8)))  float  v8f;

__device__ __forceinline__ void wait_async_zero() {
#if __has_builtin(__builtin_amdgcn_s_wait_asynccnt)
    __builtin_amdgcn_s_wait_asynccnt(0);
#else
    asm volatile("s_wait_asynccnt 0x0" ::: "memory");
#endif
}

// One per-lane async 16B copy: global -> LDS (tracked by ASYNCcnt).
__device__ __forceinline__ void async_copy_b128(unsigned int lds_byte_addr,
                                                const void* gaddr) {
    unsigned long long ga = (unsigned long long)gaddr;
    asm volatile("global_load_async_to_lds_b128 %0, %1, off"
                 :: "v"(lds_byte_addr), "v"(ga)
                 : "memory");
}

// ---------------------------------------------------------------------------
// fp32 -> bf16 (round-to-nearest-even) elementwise convert
// ---------------------------------------------------------------------------
__global__ void __launch_bounds__(256)
k_f32_to_bf16(const float* __restrict__ src, unsigned short* __restrict__ dst, int n)
{
    int idx = blockIdx.x * 256 + threadIdx.x;
    if (idx < n) {
        unsigned int u = __float_as_uint(src[idx]);
        u += 0x7FFFu + ((u >> 16) & 1u);
        dst[idx] = (unsigned short)(u >> 16);
    }
}

// ---------------------------------------------------------------------------
// Tiled bf16 WMMA GEMM:  C[M,N] = A[M,K] * W[N,K]^T + bias  (optional ReLU)
//   block = 256 threads (8 waves), block tile 128(M) x 128(N)
//   wave grid 4(M) x 2(N): each wave owns 32x64 = 2x4 v_wmma_f32_16x16x32_bf16
//   A/B staged via GLOBAL_LOAD_ASYNC_TO_LDS_B128, double-buffered LDS,
//   one barrier per K-step. Fragment gathers follow ISA 7.12.2 layouts.
// Requires: M % 128 == 0, N % 128 == 0, K % 32 == 0 (true for all call sites).
// ---------------------------------------------------------------------------
__global__ void __launch_bounds__(256)
k_gemm_bf16(const unsigned short* __restrict__ A,   // [M,K] bf16 row-major
            const unsigned short* __restrict__ W,   // [N,K] bf16 row-major
            const float* __restrict__ bias,         // [N]
            float* __restrict__ C,                  // [M,N] fp32
            int M, int N, int K, int relu)
{
    __shared__ unsigned int lsA[2][128 * LDS_STRIDE];  // 128 rows x 32 bf16
    __shared__ unsigned int lsB[2][128 * LDS_STRIDE];  // 128 rows x 32 bf16

    const int tid  = threadIdx.x;
    const int lane = tid & 31;
    const int wave = tid >> 5;
    const int wm   = wave & 3;      // wave row: 0..3  -> 32 rows each
    const int wn   = wave >> 2;     // wave col: 0..1  -> 64 cols each
    const int bm   = blockIdx.y * 128;
    const int bn   = blockIdx.x * 128;

    const int m_h = lane >> 4;      // lane half (0/1)
    const int m_l = lane & 15;      // lane index within half

    // This thread's two staging segments (shared map for A and B tiles):
    // 512 x 16B segments per 128x32 tile, 2 per thread.
    const int s0   = tid;           // segment 0
    const int s1   = tid + 256;     // segment 1
    const int row0 = s0 >> 2, sc0 = s0 & 3;
    const int row1 = s1 >> 2, sc1 = s1 & 3;

    union Frag { v16bf v; unsigned int u[8]; };

    v8f acc[2][4];
    #pragma unroll
    for (int i = 0; i < 2; ++i)
        #pragma unroll
        for (int j = 0; j < 4; ++j)
            #pragma unroll
            for (int e = 0; e < 8; ++e) acc[i][j][e] = 0.0f;

    auto issue_tile = [&](int buf, int kt) {
        async_copy_b128((unsigned int)(unsigned long long)&lsA[buf][row0 * LDS_STRIDE + sc0 * 4],
                        A + (size_t)(bm + row0) * K + kt + sc0 * 8);
        async_copy_b128((unsigned int)(unsigned long long)&lsA[buf][row1 * LDS_STRIDE + sc1 * 4],
                        A + (size_t)(bm + row1) * K + kt + sc1 * 8);
        async_copy_b128((unsigned int)(unsigned long long)&lsB[buf][row0 * LDS_STRIDE + sc0 * 4],
                        W + (size_t)(bn + row0) * K + kt + sc0 * 8);
        async_copy_b128((unsigned int)(unsigned long long)&lsB[buf][row1 * LDS_STRIDE + sc1 * 4],
                        W + (size_t)(bn + row1) * K + kt + sc1 * 8);
    };

    const int nk = K >> 5;
    issue_tile(0, 0);

    int buf = 0;
    for (int it = 0; it < nk; ++it) {
        // Make this wave's async writes land, then make all waves' writes visible.
        wait_async_zero();
        __syncthreads();

        // Overlap: stage the next K-tile into the other buffer.
        if (it + 1 < nk) issue_tile(buf ^ 1, (it + 1) << 5);
        // Prefetch the tile two K-steps ahead into L2 (global_prefetch_b8).
        if (it + 2 < nk) {
            __builtin_prefetch(A + (size_t)(bm + row0) * K + ((it + 2) << 5) + sc0 * 8, 0, 3);
            __builtin_prefetch(W + (size_t)(bn + row0) * K + ((it + 2) << 5) + sc0 * 8, 0, 3);
        }

        // ---- build fragments (ISA 7.12.2 16-bit layouts) ----
        Frag afr[2], bfr[4];
        #pragma unroll
        for (int ai = 0; ai < 2; ++ai) {
            const int r = wm * 32 + ai * 16 + m_l;          // A row for this lane
            #pragma unroll
            for (int v = 0; v < 8; ++v) {
                // A 16x32: VGPR v holds K = (v<4?0:16) + half*8 + (v&3)*2
                const int k = ((v & 4) << 2) + m_h * 8 + (v & 3) * 2;
                afr[ai].u[v] = lsA[buf][r * LDS_STRIDE + (k >> 1)];
            }
        }
        #pragma unroll
        for (int bi = 0; bi < 4; ++bi) {
            const int c = wn * 64 + bi * 16 + m_l;          // B column (= W row)
            #pragma unroll
            for (int v = 0; v < 8; ++v) {
                // B 32x16: VGPR v holds K = (v<4?0:16) + half*8 + (v&3)*2
                const int k = ((v & 4) << 2) + m_h * 8 + (v & 3) * 2;
                bfr[bi].u[v] = lsB[buf][c * LDS_STRIDE + (k >> 1)];
            }
        }

        // ---- 8 WMMAs per K-step: each A-frag reused across 4 B-frags ----
        #pragma unroll
        for (int ai = 0; ai < 2; ++ai)
            #pragma unroll
            for (int bi = 0; bi < 4; ++bi)
                acc[ai][bi] = __builtin_amdgcn_wmma_f32_16x16x32_bf16(
                    false, afr[ai].v, false, bfr[bi].v,
                    (short)0, acc[ai][bi], false, false);

        buf ^= 1;
    }

    // ---- epilogue: C/D layout -> lane n = lane&15, M = r + 8*half ----
    #pragma unroll
    for (int ai = 0; ai < 2; ++ai) {
        #pragma unroll
        for (int bi = 0; bi < 4; ++bi) {
            const int n  = bn + wn * 64 + bi * 16 + m_l;
            const float bv = bias[n];
            #pragma unroll
            for (int r = 0; r < 8; ++r) {
                const int m = bm + wm * 32 + ai * 16 + r + 8 * m_h;
                float val = acc[ai][bi][r] + bv;
                if (relu) val = fmaxf(val, 0.0f);
                C[(size_t)m * N + n] = val;
            }
        }
    }
}

// ---------------------------------------------------------------------------
// Windowed attention, one wave per (b, h, q).  Streaming softmax, fp32.
// qkv: [B*L, 3D] rows; q = cols [0,512), k = [512,1024), v = [1024,1536)
// ctx: [B*L, D]
// ---------------------------------------------------------------------------
__global__ void __launch_bounds__(32)
k_attn(const float* __restrict__ qkv, const int* __restrict__ xlen,
       float* __restrict__ ctx)
{
    const int i    = blockIdx.x;
    const int h    = blockIdx.y;
    const int b    = blockIdx.z;
    const int lane = threadIdx.x;          // 0..31 (wave32)
    const int len  = xlen[b];

    const size_t rowQ = ((size_t)b * LL + i) * (3 * DD);
    const int d0 = lane, d1 = lane + 32;
    const float q0 = qkv[rowQ + h * DHH + d0];
    const float q1 = qkv[rowQ + h * DHH + d1];

    int jlo = i - WINDOW; if (jlo < 0) jlo = 0;
    int jhi = i + WINDOW; if (jhi > LL - 1) jhi = LL - 1;

    const bool irow_pad = (i >= len);
    float mmax = -3.0e38f, ssum = 0.0f, c0 = 0.0f, c1 = 0.0f;

    for (int j = jlo; j <= jhi; ++j) {
        const size_t rowB = ((size_t)b * LL + j) * (3 * DD) + h * DHH;
        float p = q0 * qkv[rowB + DD + d0] + q1 * qkv[rowB + DD + d1];
        #pragma unroll
        for (int off = 16; off; off >>= 1) p += __shfl_xor(p, off, 32);
        float s = p * 0.125f;                              // 1/sqrt(64)
        if ((irow_pad || j >= len) && (j != i)) s += NEGB; // pad mask, diag kept
        const float nm = fmaxf(mmax, s);
        const float sc = __expf(mmax - nm);
        const float pe = __expf(s - nm);
        ssum = ssum * sc + pe;
        c0 = c0 * sc + pe * qkv[rowB + 2 * DD + d0];
        c1 = c1 * sc + pe * qkv[rowB + 2 * DD + d1];
        mmax = nm;
    }
    const float inv = 1.0f / ssum;
    float* orow = ctx + ((size_t)b * LL + i) * DD + h * DHH;
    orow[d0] = c0 * inv;
    orow[d1] = c1 * inv;
}

// ---------------------------------------------------------------------------
// out = LayerNorm(x + r) * g + beta, one block (256 thr) per row of 512
// ---------------------------------------------------------------------------
__global__ void __launch_bounds__(256)
k_add_ln(const float* __restrict__ x, const float* __restrict__ r,
         const float* __restrict__ g, const float* __restrict__ beta,
         float* __restrict__ out)
{
    __shared__ float red[256];
    const int row = blockIdx.x;
    const int t   = threadIdx.x;
    const size_t base = (size_t)row * DD;

    const float y0 = x[base + t]       + r[base + t];
    const float y1 = x[base + t + 256] + r[base + t + 256];

    red[t] = y0 + y1;
    __syncthreads();
    #pragma unroll
    for (int s = 128; s; s >>= 1) { if (t < s) red[t] += red[t + s]; __syncthreads(); }
    const float mu = red[0] * (1.0f / (float)DD);
    __syncthreads();

    const float dy0 = y0 - mu, dy1 = y1 - mu;
    red[t] = dy0 * dy0 + dy1 * dy1;
    __syncthreads();
    #pragma unroll
    for (int s = 128; s; s >>= 1) { if (t < s) red[t] += red[t + s]; __syncthreads(); }
    const float rstd = rsqrtf(red[0] * (1.0f / (float)DD) + 1e-5f);

    out[base + t]       = dy0 * rstd * g[t]       + beta[t];
    out[base + t + 256] = dy1 * rstd * g[t + 256] + beta[t + 256];
}

// ---------------------------------------------------------------------------
// mean over L:  out[b,d] = (1/L) * sum_l h[b,l,d]
// ---------------------------------------------------------------------------
__global__ void __launch_bounds__(256)
k_mean(const float* __restrict__ h, float* __restrict__ out)
{
    const int idx = blockIdx.x * 256 + threadIdx.x;   // 0 .. B*D-1
    const int b = idx >> 9, d = idx & (DD - 1);
    const float* p = h + (size_t)b * LL * DD + d;
    float s = 0.0f;
    for (int l = 0; l < LL; ++l) s += p[(size_t)l * DD];
    out[idx] = s * (1.0f / (float)LL);
}

// ---------------------------------------------------------------------------
// Workspace layout (bytes, all 256-aligned)
// ---------------------------------------------------------------------------
static const size_t OFF_WQKV = 0;                                  // 4*1536*512 bf16
static const size_t OFF_WO   = OFF_WQKV + (size_t)NLAY*3*DD*DD*2;  // 4*512*512 bf16
static const size_t OFF_W1   = OFF_WO   + (size_t)NLAY*DD*DD*2;    // 4*2048*512 bf16
static const size_t OFF_W2   = OFF_W1   + (size_t)NLAY*DFFN*DD*2;  // 4*512*2048 bf16
static const size_t OFF_XBF  = OFF_W2   + (size_t)NLAY*DD*DFFN*2;  // 8192*2048 bf16
static const size_t OFF_G1   = OFF_XBF  + (size_t)MM*DFFN*2;       // 8192*2048 fp32
static const size_t OFF_A1   = OFF_G1   + (size_t)MM*DFFN*4;       // 8192*512 fp32
static const size_t OFF_HA   = OFF_A1   + (size_t)MM*DD*4;         // 8192*512 fp32
static const size_t OFF_HB   = OFF_HA   + (size_t)MM*DD*4;         // 8192*512 fp32

extern "C" void kernel_launch(void* const* d_in, const int* in_sizes, int n_in,
                              void* d_out, int out_size, void* d_ws, size_t ws_size,
                              hipStream_t stream)
{
    const float* x    = (const float*)d_in[0];
    const int*   xlen = (const int*)  d_in[1];
    const float* wqkv = (const float*)d_in[2];
    const float* bqkv = (const float*)d_in[3];
    const float* wo   = (const float*)d_in[4];
    const float* bo   = (const float*)d_in[5];
    const float* w1   = (const float*)d_in[6];
    const float* b1   = (const float*)d_in[7];
    const float* w2   = (const float*)d_in[8];
    const float* b2   = (const float*)d_in[9];
    const float* ln1g = (const float*)d_in[10];
    const float* ln1b = (const float*)d_in[11];
    const float* ln2g = (const float*)d_in[12];
    const float* ln2b = (const float*)d_in[13];
    float* out = (float*)d_out;

    char* ws = (char*)d_ws;
    unsigned short* wqkv_bf = (unsigned short*)(ws + OFF_WQKV);
    unsigned short* wo_bf   = (unsigned short*)(ws + OFF_WO);
    unsigned short* w1_bf   = (unsigned short*)(ws + OFF_W1);
    unsigned short* w2_bf   = (unsigned short*)(ws + OFF_W2);
    unsigned short* xbf     = (unsigned short*)(ws + OFF_XBF);
    float* G1 = (float*)(ws + OFF_G1);
    float* A1 = (float*)(ws + OFF_A1);
    float* hA = (float*)(ws + OFF_HA);
    float* hB = (float*)(ws + OFF_HB);

    auto cvt = [&](const float* s, unsigned short* d, int n) {
        k_f32_to_bf16<<<(n + 255) / 256, 256, 0, stream>>>(s, d, n);
    };
    auto gemm = [&](const unsigned short* Abf, const unsigned short* Wbf,
                    const float* bias, float* Cout, int M, int N, int K, int relu) {
        dim3 g(N / 128, M / 128);
        k_gemm_bf16<<<g, 256, 0, stream>>>(Abf, Wbf, bias, Cout, M, N, K, relu);
    };

    // ---- weights -> bf16 (once per call; deterministic) ----
    cvt(wqkv, wqkv_bf, NLAY * 3 * DD * DD);
    cvt(wo,   wo_bf,   NLAY * DD * DD);
    cvt(w1,   w1_bf,   NLAY * DFFN * DD);
    cvt(w2,   w2_bf,   NLAY * DD * DFFN);

    // ---- h = x ----
    hipMemcpyAsync(hA, x, (size_t)MM * DD * sizeof(float),
                   hipMemcpyDeviceToDevice, stream);

    for (int i = 0; i < NLAY; ++i) {
        const unsigned short* Wqkv = wqkv_bf + (size_t)i * 3 * DD * DD;
        const unsigned short* Wo   = wo_bf   + (size_t)i * DD * DD;
        const unsigned short* W1   = w1_bf   + (size_t)i * DFFN * DD;
        const unsigned short* W2   = w2_bf   + (size_t)i * DD * DFFN;

        // QKV projection: [8192,512] x [1536,512]^T -> [8192,1536]
        cvt(hA, xbf, MM * DD);
        gemm(xbf, Wqkv, bqkv + (size_t)i * 3 * DD, G1, MM, 3 * DD, DD, 0);

        // windowed attention -> ctx [8192,512]
        k_attn<<<dim3(LL, HH, BB), 32, 0, stream>>>(G1, xlen, A1);

        // output projection
        cvt(A1, xbf, MM * DD);
        gemm(xbf, Wo, bo + (size_t)i * DD, A1, MM, DD, DD, 0);

        // h = LN(h + attn_out)
        k_add_ln<<<MM, 256, 0, stream>>>(hA, A1,
                                         ln1g + (size_t)i * DD, ln1b + (size_t)i * DD, hB);

        // FFN1 (+ReLU): [8192,512] x [2048,512]^T -> [8192,2048]
        cvt(hB, xbf, MM * DD);
        gemm(xbf, W1, b1 + (size_t)i * DFFN, G1, MM, DFFN, DD, 1);

        // FFN2: [8192,2048] x [512,2048]^T -> [8192,512]
        cvt(G1, xbf, MM * DFFN);
        gemm(xbf, W2, b2 + (size_t)i * DD, A1, MM, DD, DFFN, 0);

        // h = LN(h + ffn_out)  (ping-pong back to hA)
        k_add_ln<<<MM, 256, 0, stream>>>(hB, A1,
                                         ln2g + (size_t)i * DD, ln2b + (size_t)i * DD, hA);
    }

    // mean pool over L -> [B, D]
    k_mean<<<(BB * DD) / 256, 256, 0, stream>>>(hA, out);
}
